// AttentionLstm_33775622816014
// MI455X (gfx1250) — matrix operand order repaired and checked by reading, not compiled, not run on hardware
//
#include <hip/hip_runtime.h>
#include <math.h>

// ---------------------------------------------------------------------------
// AttentionLSTM decoder forward for gfx1250 (MI455X).
// B=64, T=20, L=49, F=2048, E=512, H=512, V=30000.
// All GEMMs run as bf16 WMMA (v_wmma_f32_16x16x32_bf16) with f32 accumulate.
// Weights are converted to bf16 once per call into d_ws; fc_W (30.7MB bf16)
// stays resident in the 192MB L2 across the 20 sequential timesteps.
// ---------------------------------------------------------------------------

typedef __bf16 bf16;
typedef __attribute__((ext_vector_type(16))) __bf16 v16bf;
typedef __attribute__((ext_vector_type(8)))  __bf16 v8bf;
typedef __attribute__((ext_vector_type(8)))  float  v8f;

#define Bc 64
#define Tc 20
#define Lc 49
#define Fc 2048
#define Ec 512
#define Hc 512
#define Vc 30000
#define Vpad 30016  // fc_W bf16 buffer padded to a multiple of 64 columns
#define KCAT 3072   // E+F+H : concatenated K for fused [W_ih|W_hh] gates GEMM

// ---------------------------------------------------------------------------
// Skinny WMMA GEMM:  C[M,N] = A[M,K] * W[N,K]^T + bias[N]
// A row stride = lda (elements), W row stride = K, C row stride = ldc.
// One wave computes a 32x64 tile: 2 M-tiles x 4 N-tiles = 8 independent
// accumulator chains -> WMMA->WMMA RAW distance 8 (>= 5 required by ISA
// 7.12.1, so no hazard NOPs), and each B fragment feeds 2 WMMAs.
// Tile indices are forced wave-uniform via readfirstlane so all control flow
// is scalar (EXEC stays all-ones through every WMMA, no divergent phis).
// The K-loop is guard-free: W must be padded to ceil(N/64)*64 rows; only the
// epilogue stores are guarded (scalar branch per column tile).
// A and B both use the same contiguous k order (k = 16*half + s), so both fragments are
// contiguous global loads (no LDS shuffle stage needed).
// Requires: M % 32 == 0, N % 16 == 0, K % 32 == 0.
// ---------------------------------------------------------------------------
__global__ __launch_bounds__(256)
void wmma_gemm_nt(const bf16* __restrict__ A, int lda,
                  const bf16* __restrict__ W,
                  const float* __restrict__ bias,
                  float* __restrict__ C, long long ldc,
                  int M, int N, int K)
{
    const int lane = threadIdx.x & 31;
    const int wid  = __builtin_amdgcn_readfirstlane(
                        (blockIdx.x * blockDim.x + threadIdx.x) >> 5);
    const int tiles_m2 = M >> 5;                  // 32-row supertiles
    const int tiles_n  = N >> 4;
    const int groups   = (tiles_n + 3) >> 2;
    if (wid >= tiles_m2 * groups) return;         // scalar exit

    const int tm   = wid % tiles_m2;              // SGPR tile coords
    const int gn   = wid / tiles_m2;
    const int n0   = gn * 64;
    const int half = lane >> 4;                   // K-half selector (VGPR)
    const int l15  = lane & 15;

    // A fragment pointers: two 16-row tiles, two 8-elem chunks per K-step
    const bf16* Ap0 = A + (size_t)(tm * 32 +      l15) * (size_t)lda;
    const bf16* Ap1 = A + (size_t)(tm * 32 + 16 + l15) * (size_t)lda;

    // B fragment pointers: lane l15 = column within tile, 16 contiguous K vals
    const bf16* Bp0 = W + (size_t)(n0 +  0 + l15) * (size_t)K + (size_t)(half * 16);
    const bf16* Bp1 = W + (size_t)(n0 + 16 + l15) * (size_t)K + (size_t)(half * 16);
    const bf16* Bp2 = W + (size_t)(n0 + 32 + l15) * (size_t)K + (size_t)(half * 16);
    const bf16* Bp3 = W + (size_t)(n0 + 48 + l15) * (size_t)K + (size_t)(half * 16);

    v8f acc00 = {}, acc01 = {}, acc02 = {}, acc03 = {};
    v8f acc10 = {}, acc11 = {}, acc12 = {}, acc13 = {};

    for (int k0 = 0; k0 < K; k0 += 32) {
        // A fragments (16x32 bf16): lanes 0-15 rows M, K=0..15;
        // lanes 16-31 same rows, K=16..31.
        v8bf a0lo = *(const v8bf*)(Ap0 + k0 + half * 16);
        v8bf a0hi = *(const v8bf*)(Ap0 + k0 + 8 + half * 16);
        v8bf a1lo = *(const v8bf*)(Ap1 + k0 + half * 16);
        v8bf a1hi = *(const v8bf*)(Ap1 + k0 + 8 + half * 16);
        v16bf a0 = __builtin_shufflevector(a0lo, a0hi,
                     0,1,2,3,4,5,6,7,8,9,10,11,12,13,14,15);
        v16bf a1 = __builtin_shufflevector(a1lo, a1hi,
                     0,1,2,3,4,5,6,7,8,9,10,11,12,13,14,15);

        if (k0 + 32 < K) {
            __builtin_prefetch(Ap0 + k0 + 32, 0, 1);   // global_prefetch_b8
            __builtin_prefetch(Bp0 + k0 + 32, 0, 1);
        }

        v16bf b0 = *(const v16bf*)(Bp0 + k0);
        v16bf b1 = *(const v16bf*)(Bp1 + k0);
        v16bf b2 = *(const v16bf*)(Bp2 + k0);
        v16bf b3 = *(const v16bf*)(Bp3 + k0);

        // 8 independent WMMA chains: B fragment reused across both M-tiles.
        acc00 = __builtin_amdgcn_wmma_f32_16x16x32_bf16(
                    false, a0, false, b0, (short)0, acc00, false, false);
        acc10 = __builtin_amdgcn_wmma_f32_16x16x32_bf16(
                    false, a1, false, b0, (short)0, acc10, false, false);
        acc01 = __builtin_amdgcn_wmma_f32_16x16x32_bf16(
                    false, a0, false, b1, (short)0, acc01, false, false);
        acc11 = __builtin_amdgcn_wmma_f32_16x16x32_bf16(
                    false, a1, false, b1, (short)0, acc11, false, false);
        acc02 = __builtin_amdgcn_wmma_f32_16x16x32_bf16(
                    false, a0, false, b2, (short)0, acc02, false, false);
        acc12 = __builtin_amdgcn_wmma_f32_16x16x32_bf16(
                    false, a1, false, b2, (short)0, acc12, false, false);
        acc03 = __builtin_amdgcn_wmma_f32_16x16x32_bf16(
                    false, a0, false, b3, (short)0, acc03, false, false);
        acc13 = __builtin_amdgcn_wmma_f32_16x16x32_bf16(
                    false, a1, false, b3, (short)0, acc13, false, false);
    }

    // Epilogue: C/D layout — VGPR r holds row (r + 8*half), col = lane&15.
    v8f accs[2][4] = {{acc00, acc01, acc02, acc03},
                      {acc10, acc11, acc12, acc13}};
#pragma unroll
    for (int j = 0; j < 4; ++j) {
        int ncol = n0 + j * 16;                    // scalar
        if (ncol < N) {                            // scalar store guard
            int col = ncol + l15;
            float bv = bias ? bias[col] : 0.0f;
#pragma unroll
            for (int mi = 0; mi < 2; ++mi) {
#pragma unroll
                for (int r = 0; r < 8; ++r) {
                    int row = tm * 32 + mi * 16 + half * 8 + r;
                    C[(long long)row * ldc + col] = accs[mi][j][r] + bv;
                }
            }
        }
    }
}

// ---------------------------------------------------------------------------
// f32 -> bf16 strided row copy (weight conversion / concatenation)
// ---------------------------------------------------------------------------
__global__ __launch_bounds__(256)
void cvt_rows(const float* __restrict__ src, bf16* __restrict__ dst,
              int rows, int cols, int sstride, int dstride, int doff)
{
    long long idx = (long long)blockIdx.x * blockDim.x + threadIdx.x;
    long long total = (long long)rows * cols;
    if (idx >= total) return;
    int r = (int)(idx / cols), c = (int)(idx % cols);
    dst[(size_t)r * dstride + doff + c] = (bf16)src[(size_t)r * sstride + c];
}

// zero-fill bf16 region (fc_W padding rows)
__global__ __launch_bounds__(256)
void fill_zero_bf16(bf16* __restrict__ dst, long long n)
{
    long long i = (long long)blockIdx.x * blockDim.x + threadIdx.x;
    if (i < n) dst[i] = (bf16)0.0f;
}

// Teacher-forced embedding gather: we[b,t,:] = (t==0) ? 0 : emb[cap[b,t-1]]
__global__ __launch_bounds__(256)
void embed_shift(const int* __restrict__ cap, const float* __restrict__ emb,
                 bf16* __restrict__ we)
{
    int idx = blockIdx.x * blockDim.x + threadIdx.x;
    if (idx >= Bc * Tc * Ec) return;
    int b = idx / (Tc * Ec);
    int r = idx % (Tc * Ec);
    int t = r / Ec, e = r % Ec;
    float v = 0.0f;
    if (t > 0) v = emb[(size_t)cap[b * Tc + t - 1] * Ec + e];
    we[idx] = (bf16)v;
}

// mean over L of features -> mean[b,f]
__global__ __launch_bounds__(256)
void mean_feat(const float* __restrict__ feat, float* __restrict__ mean)
{
    int idx = blockIdx.x * blockDim.x + threadIdx.x;
    if (idx >= Bc * Fc) return;
    int b = idx / Fc, f = idx % Fc;
    float s = 0.0f;
    for (int l = 0; l < Lc; ++l)
        s += feat[((size_t)b * Lc + l) * Fc + f];
    mean[idx] = s * (1.0f / 49.0f);
}

// h0 = mean@ih_W.T + ih_b (bf16 into xh h-slot), c0 = mean@ic_W.T + ic_b
__global__ __launch_bounds__(256)
void init_hc(const float* __restrict__ mean,
             const float* __restrict__ ihW, const float* __restrict__ ihb,
             const float* __restrict__ icW, const float* __restrict__ icb,
             bf16* __restrict__ xh, float* __restrict__ c)
{
    int idx = blockIdx.x * blockDim.x + threadIdx.x;
    if (idx >= Bc * Hc) return;
    int b = idx / Hc, j = idx % Hc;
    const float* m  = mean + (size_t)b * Fc;
    const float* wh = ihW + (size_t)j * Fc;
    const float* wc = icW + (size_t)j * Fc;
    float ah = ihb[j], ac = icb[j];
    for (int f = 0; f < Fc; ++f) { ah += m[f] * wh[f]; ac += m[f] * wc[f]; }
    xh[(size_t)b * KCAT + (Ec + Fc) + j] = (bf16)ah;
    c[idx] = ac;
}

// combined gate bias = b_ih + b_hh
__global__ __launch_bounds__(256)
void add_bias(const float* __restrict__ a, const float* __restrict__ b,
              float* __restrict__ o, int n)
{
    int i = blockIdx.x * blockDim.x + threadIdx.x;
    if (i < n) o[i] = a[i] + b[i];
}

// ---------------------------------------------------------------------------
// Fused attention: score = tanh(a1 + a2)@va + bv ; softmax ; context = w@feat
// One block per batch element. Writes w into the second output tensor and
// (context | word-embed) bf16 into the gates-GEMM input xh.
// ---------------------------------------------------------------------------
__global__ __launch_bounds__(256)
void attn_ctx(const float* __restrict__ a1, const float* __restrict__ a2,
              const float* __restrict__ va, const float* __restrict__ bv,
              const float* __restrict__ feat, const bf16* __restrict__ we,
              bf16* __restrict__ xh, float* __restrict__ wout, int t)
{
    __shared__ float s_sc[64];
    __shared__ float s_w[64];
    const int b    = blockIdx.x;
    const int lane = threadIdx.x & 31;
    const int wv   = threadIdx.x >> 5;

    // scores: 8 waves x (Lc/8) rows, wave-reduce over H
    for (int l = wv; l < Lc; l += 8) {
        const float* a1p = a1 + ((size_t)b * Lc + l) * Hc;
        const float* a2p = a2 + (size_t)b * Hc;
        float acc = 0.0f;
        for (int h = lane; h < Hc; h += 32)
            acc += tanhf(a1p[h] + a2p[h]) * va[h];
#pragma unroll
        for (int off = 16; off > 0; off >>= 1)
            acc += __shfl_down(acc, off, 32);
        if (lane == 0) s_sc[l] = acc + bv[0];
    }
    __syncthreads();

    if (threadIdx.x == 0) {               // softmax over 49 (trivial size)
        float m = -3.0e38f;
        for (int l = 0; l < Lc; ++l) m = fmaxf(m, s_sc[l]);
        float s = 0.0f;
        for (int l = 0; l < Lc; ++l) { float e = expf(s_sc[l] - m); s_w[l] = e; s += e; }
        float inv = 1.0f / s;
        for (int l = 0; l < Lc; ++l) s_w[l] *= inv;
    }
    __syncthreads();

    if (threadIdx.x < Lc)
        wout[((size_t)b * Tc + t) * Lc + threadIdx.x] = s_w[threadIdx.x];

    // context[b,f] = sum_l w[l] * feat[b,l,f]  -> xh[b, E + f] (bf16)
    for (int f = threadIdx.x; f < Fc; f += 256) {
        const float* fp = feat + (size_t)b * Lc * Fc + f;
        float acc = 0.0f;
        for (int l = 0; l < Lc; ++l) acc += s_w[l] * fp[(size_t)l * Fc];
        xh[(size_t)b * KCAT + Ec + f] = (bf16)acc;
    }
    // word embed slot: xh[b, 0:E] = we[b,t,:]
    for (int e = threadIdx.x; e < Ec; e += 256)
        xh[(size_t)b * KCAT + e] = we[((size_t)b * Tc + t) * Ec + e];
}

// LSTM pointwise update: h2 (bf16 into xh h-slot), c updated in place
__global__ __launch_bounds__(256)
void lstm_update(const float* __restrict__ gates, float* __restrict__ c,
                 bf16* __restrict__ xh)
{
    int idx = blockIdx.x * blockDim.x + threadIdx.x;
    if (idx >= Bc * Hc) return;
    int b = idx / Hc, j = idx % Hc;
    const float* g = gates + (size_t)b * (4 * Hc);
    float gi = 1.0f / (1.0f + expf(-g[j]));
    float gf = 1.0f / (1.0f + expf(-g[Hc + j]));
    float gg = tanhf(g[2 * Hc + j]);
    float go = 1.0f / (1.0f + expf(-g[3 * Hc + j]));
    float c2 = gf * c[idx] + gi * gg;
    c[idx] = c2;
    xh[(size_t)b * KCAT + (Ec + Fc) + j] = (bf16)(go * tanhf(c2));
}

// ---------------------------------------------------------------------------
// Host-side launch helpers
// ---------------------------------------------------------------------------
static void gemm(const bf16* A, int lda, const bf16* W, const float* bias,
                 float* C, long long ldc, int M, int N, int K, hipStream_t s)
{
    int tiles_m2 = M >> 5;                  // 32-row supertiles
    int groups   = ((N >> 4) + 3) >> 2;
    int waves    = tiles_m2 * groups;
    int blocks   = (waves + 7) / 8;         // 8 waves (256 threads) per block
    wmma_gemm_nt<<<blocks, 256, 0, s>>>(A, lda, W, bias, C, ldc, M, N, K);
}

extern "C" void kernel_launch(void* const* d_in, const int* in_sizes, int n_in,
                              void* d_out, int out_size, void* d_ws, size_t ws_size,
                              hipStream_t stream)
{
    const int*   captions = (const int*)  d_in[0];
    const float* features = (const float*)d_in[1];
    const float* emb      = (const float*)d_in[2];
    const float* W_ih     = (const float*)d_in[3];
    const float* W_hh     = (const float*)d_in[4];
    const float* b_ih     = (const float*)d_in[5];
    const float* b_hh     = (const float*)d_in[6];
    const float* fc_W     = (const float*)d_in[7];
    const float* fc_b     = (const float*)d_in[8];
    const float* Wa       = (const float*)d_in[9];
    const float* ba       = (const float*)d_in[10];
    const float* Ua       = (const float*)d_in[11];
    const float* bu       = (const float*)d_in[12];
    const float* va       = (const float*)d_in[13];
    const float* bv       = (const float*)d_in[14];
    const float* ih_W     = (const float*)d_in[15];
    const float* ih_b     = (const float*)d_in[16];
    const float* ic_W     = (const float*)d_in[17];
    const float* ic_b     = (const float*)d_in[18];

    float* out  = (float*)d_out;                        // [B,T,V]
    float* wout = out + (size_t)Bc * Tc * Vc;           // [B,T,L]

    // ---- workspace carve-up (~69 MB total, 256B aligned) ----
    char*  ws  = (char*)d_ws;
    size_t off = 0;
    auto take = [&](size_t bytes) {
        size_t r = off; off += (bytes + 255) & ~(size_t)255; return r;
    };
    bf16*  fcW_bf  = (bf16*)(ws + take((size_t)Vpad * Hc * 2));      // 30.7 MB
    bf16*  Wcat_bf = (bf16*)(ws + take((size_t)(4*Hc) * KCAT * 2));  // 12.6 MB
    bf16*  Ua_bf   = (bf16*)(ws + take((size_t)Hc * Hc * 2));
    bf16*  Wa_bf   = (bf16*)(ws + take((size_t)Hc * Fc * 2));
    bf16*  feat_bf = (bf16*)(ws + take((size_t)Bc * Lc * Fc * 2));   // 12.8 MB
    bf16*  we_bf   = (bf16*)(ws + take((size_t)Bc * Tc * Ec * 2));
    float* a1      = (float*)(ws + take((size_t)Bc * Lc * Hc * 4));  //  6.4 MB
    float* a2      = (float*)(ws + take((size_t)Bc * Hc * 4));
    bf16*  xh      = (bf16*) (ws + take((size_t)Bc * KCAT * 2));     // [we|ctx|h]
    float* cbuf    = (float*)(ws + take((size_t)Bc * Hc * 4));
    float* gates   = (float*)(ws + take((size_t)Bc * 4 * Hc * 4));
    float* meanf   = (float*)(ws + take((size_t)Bc * Fc * 4));
    float* gbias   = (float*)(ws + take((size_t)4 * Hc * 4));
    (void)ws_size; (void)in_sizes; (void)n_in; (void)out_size;

    auto grid1d = [](long long n) { return (int)((n + 255) / 256); };

    // ---- one-time setup: bf16 weight conversion ----
    cvt_rows<<<grid1d((long long)Vc * Hc), 256, 0, stream>>>(
        fc_W, fcW_bf, Vc, Hc, Hc, Hc, 0);
    fill_zero_bf16<<<grid1d((long long)(Vpad - Vc) * Hc), 256, 0, stream>>>(
        fcW_bf + (size_t)Vc * Hc, (long long)(Vpad - Vc) * Hc);     // pad rows
    cvt_rows<<<grid1d((long long)(4*Hc) * (Ec+Fc)), 256, 0, stream>>>(
        W_ih, Wcat_bf, 4*Hc, Ec+Fc, Ec+Fc, KCAT, 0);                // K 0..2559
    cvt_rows<<<grid1d((long long)(4*Hc) * Hc), 256, 0, stream>>>(
        W_hh, Wcat_bf, 4*Hc, Hc, Hc, KCAT, Ec+Fc);                  // K 2560..3071
    cvt_rows<<<grid1d((long long)Hc * Hc), 256, 0, stream>>>(
        Ua, Ua_bf, Hc, Hc, Hc, Hc, 0);
    cvt_rows<<<grid1d((long long)Hc * Fc), 256, 0, stream>>>(
        Wa, Wa_bf, Hc, Fc, Fc, Fc, 0);
    cvt_rows<<<grid1d((long long)Bc * Lc * Fc), 256, 0, stream>>>(
        features, feat_bf, Bc * Lc, Fc, Fc, Fc, 0);

    embed_shift<<<grid1d((long long)Bc * Tc * Ec), 256, 0, stream>>>(
        captions, emb, we_bf);
    mean_feat<<<grid1d((long long)Bc * Fc), 256, 0, stream>>>(features, meanf);
    init_hc<<<grid1d((long long)Bc * Hc), 256, 0, stream>>>(
        meanf, ih_W, ih_b, ic_W, ic_b, xh, cbuf);
    add_bias<<<grid1d(4 * Hc), 256, 0, stream>>>(b_ih, b_hh, gbias, 4 * Hc);

    // a1[b,l,h] = features @ Wa^T + ba   (M=3136, N=512, K=2048)
    gemm(feat_bf, Fc, Wa_bf, ba, a1, Hc, Bc * Lc, Hc, Fc, stream);

    // ---- sequential decode ----
    for (int t = 0; t < Tc; ++t) {
        // a2 = h @ Ua^T + bu   (A = h-slot of xh, lda = KCAT)
        gemm(xh + (Ec + Fc), KCAT, Ua_bf, bu, a2, Hc, Bc, Hc, Hc, stream);

        attn_ctx<<<Bc, 256, 0, stream>>>(a1, a2, va, bv, features, we_bf,
                                         xh, wout, t);

        // gates = [we|ctx|h] @ [W_ih|W_hh]^T + (b_ih + b_hh)
        gemm(xh, KCAT, Wcat_bf, gbias, gates, 4 * Hc, Bc, 4 * Hc, KCAT, stream);

        lstm_update<<<grid1d((long long)Bc * Hc), 256, 0, stream>>>(
            gates, cbuf, xh);

        // out[:,t,:] = h2 @ fc_W^T + fc_b   (N=30000, L2-resident weights)
        gemm(xh + (Ec + Fc), KCAT, fcW_bf, fc_b,
             out + (size_t)t * Vc, (long long)Tc * Vc, Bc, Vc, Hc, stream);
    }
}
